// MGraphTransformer_17669495456072
// MI455X (gfx1250) — compile-verified, matrix operand order checked by reading, not run on hardware
//
#include <hip/hip_runtime.h>
#include <cstdint>
#include <cstddef>

// ---------------------------------------------------------------------------
// Problem constants (match reference)
// ---------------------------------------------------------------------------
#define BB   128
#define NN   256
#define EE   2048
#define NV   (BB * NN)     // 32768
#define NE   (BB * EE)     // 262144
#define DV   128
#define DE   64
#define DG   128
#define DEG  32
#define NH   4
#define DH   32

typedef __attribute__((ext_vector_type(16))) _Float16 v16h;
typedef __attribute__((ext_vector_type(8)))  float    v8f;

struct GemmArgs {
    const float* A;      // (M, K) row-major
    const float* W;      // (K, N) row-major
    const float* bias;   // (N) or nullptr
    float*       out;    // (M, N) row-major
    int M, K, N;
};

__device__ __forceinline__ float dev_gelu(float x) {
    return 0.5f * x * (1.0f + erff(x * 0.70710678118654752f));
}
__device__ __forceinline__ float dev_sig(float x) { return 1.0f / (1.0f + expf(-x)); }

// CDNA5 16-bit A-fragment K permutation (swap bits 3 and 4; involution).
// Storing LDS tiles in this order makes each lane's 16 fragment halves one
// contiguous 32-byte run -> 2x ds_load_b128 instead of 16x ds_load_u16.
__device__ __forceinline__ int kperm(int p) {
    return (p & 7) | ((p & 8) << 1) | ((p & 16) >> 1);
}

// ---------------------------------------------------------------------------
// WMMA GEMM:  out = act(A @ W + bias)
//   block: 256 threads = 8 waves; tile 128 rows x 16 cols; K chunks of 32.
//   A staged f32->f16 in LDS (fragment order), W staged f16 transposed.
//   Each wave computes one 16x16 C tile with v_wmma_f32_16x16x32_f16.
// ---------------------------------------------------------------------------
template <int ACT>
__global__ void __launch_bounds__(256) k_gemm(GemmArgs g) {
    __shared__ alignas(32) _Float16 As[128 * 32];   // [row][kperm]
    __shared__ alignas(32) _Float16 Wt[16 * 32];    // [col][kperm]

    const int tid  = threadIdx.x;
    const int wave = tid >> 5;
    const int lane = tid & 31;
    const int half = lane >> 4;
    const int l16  = lane & 15;
    const int wrow = wave << 4;

    v8f acc = {};

    for (int k0 = 0; k0 < g.K; k0 += 32) {
        // --- stage A tile: 128 rows x 32 cols, dest-contiguous ------------
        {
            const int    r     = tid >> 1;
            const int    cbase = (tid & 1) << 4;
            const long   grow  = (long)blockIdx.x * 128 + r;
            const float* src   = g.A + grow * (long)g.K + k0;
            if (grow < (long)g.M && (k0 + 32) <= g.K) {
                // full tile: unguarded contiguous 8-float runs
#pragma unroll
                for (int j = 0; j < 16; ++j)
                    As[r * 32 + cbase + j] = (_Float16)src[kperm(cbase + j)];
            } else {
                const bool rok = grow < (long)g.M;
#pragma unroll
                for (int j = 0; j < 16; ++j) {
                    int   kk = kperm(cbase + j);
                    float v  = (rok && (k0 + kk) < g.K) ? src[kk] : 0.0f;
                    As[r * 32 + cbase + j] = (_Float16)v;
                }
            }
        }
        // --- stage W tile transposed: Wt[col][kperm] ----------------------
        {
            int i0 = tid * 2;                              // 512 entries, 2/thread
#pragma unroll
            for (int u = 0; u < 2; ++u) {
                int   i  = i0 + u;
                int   nn = i >> 5, p = i & 31;
                int   gk = k0 + kperm(p);
                long  gn = (long)blockIdx.y * 16 + nn;
                float v  = (gk < g.K && gn < (long)g.N) ? g.W[(long)gk * g.N + gn] : 0.0f;
                Wt[nn * 32 + p] = (_Float16)v;
            }
        }
        __syncthreads();

        // --- fragments: contiguous 32B LDS runs ---------------------------
        v16h af = *(const v16h*)(As + (wrow + l16) * 32 + half * 16);
        v16h bf = *(const v16h*)(Wt + l16 * 32 + half * 16);
        acc = __builtin_amdgcn_wmma_f32_16x16x32_f16(
            false, af, false, bf, (short)0, acc, false, false);
        __syncthreads();
    }

    // --- epilogue: C layout VGPR r -> M = r + 8*half, N = lane%16 ---------
    const long col = (long)blockIdx.y * 16 + l16;
#pragma unroll
    for (int r = 0; r < 8; ++r) {
        long row = (long)blockIdx.x * 128 + wrow + r + 8 * half;
        if (row < (long)g.M && col < (long)g.N) {
            float v = acc[r];
            if (g.bias) v += g.bias[col];
            if (ACT == 1) v = dev_gelu(v);
            g.out[row * g.N + col] = v;
        }
    }
}

// ---------------------------------------------------------------------------
// Gather / concat kernels (fully coalesced, float4)
// ---------------------------------------------------------------------------
__device__ __forceinline__ float4 ld4(const float* p) { return *(const float4*)p; }

// msg_in[e] = [ fv_mod[src[e]] (128) | fe[e] (64) ]   -> (NE, 192)
__global__ void k_cat_msg(const float* __restrict__ fv_mod, const float* __restrict__ fe,
                          const int* __restrict__ src, float* __restrict__ out, int total) {
    int i = blockIdx.x * blockDim.x + threadIdx.x;     // total = NE*48 float4 groups
    if (i >= total) return;
    int e = i / 48, c = (i % 48) * 4;
    float4 v = (c < 128) ? ld4(fv_mod + (long)src[e] * DV + c)
                         : ld4(fe + (long)e * DE + (c - 128));
    *(float4*)(out + (long)e * 192 + c) = v;
}

// u1_in[e] = [ a1[src[e]]*a2[dst[e]] (64) | fe_geo[e] (32) | fg[e/E] (128) ] -> (NE,224)
__global__ void k_cat_u1(const float* __restrict__ a1, const float* __restrict__ a2,
                         const float* __restrict__ fe_geo, const float* __restrict__ fg,
                         const int* __restrict__ src, const int* __restrict__ dst,
                         float* __restrict__ out, int total) {
    int i = blockIdx.x * blockDim.x + threadIdx.x;     // total = NE*56 float4 groups
    if (i >= total) return;
    int e = i / 56, c = (i % 56) * 4;
    float4 v;
    if (c < 64) {
        float4 x = ld4(a1 + (long)src[e] * DE + c);
        float4 y = ld4(a2 + (long)dst[e] * DE + c);
        v = make_float4(x.x * y.x, x.y * y.y, x.z * y.z, x.w * y.w);
    } else if (c < 96) {
        v = ld4(fe_geo + (long)e * DEG + (c - 64));
    } else {
        v = ld4(fg + (long)(e / EE) * DG + (c - 96));
    }
    *(float4*)(out + (long)e * 224 + c) = v;
}

// xh_in[b] = [ fe2g (64) | fv2g (128) | geo2g (64) | 256.0 ]   -> (B, 257)
__global__ void k_cat_xh(const float* __restrict__ fe2g, const float* __restrict__ fv2g,
                         const float* __restrict__ geo2g, float* __restrict__ out, int total) {
    int i = blockIdx.x * blockDim.x + threadIdx.x;     // total = BB*257
    if (i >= total) return;
    int b = i / 257, c = i % 257;
    float v;
    if (c < 64)        v = fe2g[b * 64 + c];
    else if (c < 192)  v = fv2g[b * 128 + (c - 64)];
    else if (c < 256)  v = geo2g[b * 64 + (c - 192)];
    else               v = (float)NN;
    out[i] = v;
}

// ---------------------------------------------------------------------------
// Elementwise / reduction kernels
// ---------------------------------------------------------------------------
__global__ void k_dist(const float* __restrict__ pos, float* __restrict__ dist, int n) {
    int i = blockIdx.x * blockDim.x + threadIdx.x;
    if (i >= n) return;
    float x = pos[i * 3 + 0], y = pos[i * 3 + 1], z = pos[i * 3 + 2];
    dist[i] = sqrtf(x * x + y * y + z * z);
}

// pos_diff, fe_dist, and geo_in[e] = [fe_dist, fv_dist[src], fv_dist[dst]] (NE,3)
__global__ void k_edge_pre(const float* __restrict__ pos, const float* __restrict__ vdist,
                           const int* __restrict__ src, const int* __restrict__ dst,
                           float* __restrict__ pdiff, float* __restrict__ edist,
                           float* __restrict__ geo_in, int ne) {
    int e = blockIdx.x * blockDim.x + threadIdx.x;
    if (e >= ne) return;
    int s = src[e], d = dst[e];
    float dx = pos[s * 3 + 0] - pos[d * 3 + 0];
    float dy = pos[s * 3 + 1] - pos[d * 3 + 1];
    float dz = pos[s * 3 + 2] - pos[d * 3 + 2];
    pdiff[e * 3 + 0] = dx; pdiff[e * 3 + 1] = dy; pdiff[e * 3 + 2] = dz;
    float dd = sqrtf(dx * dx + dy * dy + dz * dz);
    edist[e] = dd;
    geo_in[e * 3 + 0] = dd;
    geo_in[e * 3 + 1] = vdist[s];
    geo_in[e * 3 + 2] = vdist[d];
}

// fv_mod = fv * (1 + scale[b]) + shift[b]   (g2v = [scale|shift], per batch)
__global__ void k_film(const float* __restrict__ fv, const float* __restrict__ g2v,
                       float* __restrict__ out, int total) {
    int i = blockIdx.x * blockDim.x + threadIdx.x;
    if (i >= total) return;
    int r = i / DV, c = i % DV, b = r / NN;
    out[i] = fv[i] * (1.0f + g2v[b * 256 + c]) + g2v[b * 256 + 128 + c];
}

// fe_u = gelu(scale[dst]*msg + shift[dst]) * (gate_geo + gate_g); atomic agg
__global__ void k_msg_agg(const float* __restrict__ msg, const float* __restrict__ vu,
                          const float* __restrict__ ggeo, const float* __restrict__ gg,
                          const int* __restrict__ dst, float* __restrict__ agg, int total) {
    int i = blockIdx.x * blockDim.x + threadIdx.x;
    if (i >= total) return;
    int e = i / DV, c = i % DV, h = c / DH;
    int d = dst[e];
    float v = dev_gelu(vu[(long)d * 256 + c] * msg[i] + vu[(long)d * 256 + 128 + c]);
    v *= (ggeo[e * NH + h] + gg[(e / EE) * NH + h]);
    unsafeAtomicAdd(&agg[(long)d * DV + c], v);
}

// fe_out = fe_mid * (fe_scale + 1) + fe_shift ;  h = [shift(64)|scale(64)]
__global__ void k_fe_out(const float* __restrict__ femid, const float* __restrict__ h,
                         float* __restrict__ out, int total) {
    int i = blockIdx.x * blockDim.x + threadIdx.x;
    if (i >= total) return;
    int e = i / DE, c = i % DE;
    out[i] = femid[i] * (h[(long)e * 128 + 64 + c] + 1.0f) + h[(long)e * 128 + c];
}

// column-wise mean/min/max over middle dim: x (B, cnt, D) -> red (B, 3*D)
__global__ void k_colstats(const float* __restrict__ x, int cnt, int D,
                           float* __restrict__ red, int total) {
    int i = blockIdx.x * blockDim.x + threadIdx.x;
    if (i >= total) return;
    int b = i / D, d = i % D;
    const float* p = x + (long)b * cnt * D + d;
    float s = 0.0f, mn = 3.4e38f, mx = -3.4e38f;
    for (int j = 0; j < cnt; ++j) {
        float v = p[(long)j * D];
        s += v; mn = fminf(mn, v); mx = fmaxf(mx, v);
    }
    red[(long)b * 3 * D + d]         = s / (float)cnt;
    red[(long)b * 3 * D + D + d]     = mn;
    red[(long)b * 3 * D + 2 * D + d] = mx;
}

// concat 3 weight matrices (K,N) row-wise -> (3K,N), and sum biases
__global__ void k_cat3(const float* W0, const float* W1, const float* W2,
                       const float* b0, const float* b1, const float* b2,
                       int K, int N, float* Wc, float* bc) {
    int i = blockIdx.x * blockDim.x + threadIdx.x;
    int kn = K * N;
    if (i < 3 * kn) {
        const float* src = (i < kn) ? W0 : (i < 2 * kn) ? W1 : W2;
        Wc[i] = src[i % kn];
    }
    if (i < N) bc[i] = b0[i] + b1[i] + b2[i];
}

__global__ void k_gru(const float* __restrict__ gi, const float* __restrict__ gh,
                      const float* __restrict__ fg, float* __restrict__ out, int total) {
    int i = blockIdx.x * blockDim.x + threadIdx.x;
    if (i >= total) return;
    int b = i / DG, c = i % DG;
    const float* pi = gi + b * 3 * DG;
    const float* ph = gh + b * 3 * DG;
    float r = dev_sig(pi[c] + ph[c]);
    float z = dev_sig(pi[DG + c] + ph[DG + c]);
    float n = tanhf(pi[2 * DG + c] + r * ph[2 * DG + c]);
    out[i] = (1.0f - z) * n + z * fg[i];
}

__global__ void k_posmsg(const float* __restrict__ coef, const float* __restrict__ pdiff,
                         const float* __restrict__ edist, const int* __restrict__ dst,
                         float* __restrict__ delta, int ne) {
    int e = blockIdx.x * blockDim.x + threadIdx.x;
    if (e >= ne) return;
    float f   = coef[e];
    float inv = 1.0f / (edist[e] + 1.0f);
    int   d   = dst[e];
    for (int a = 0; a < 3; ++a)
        unsafeAtomicAdd(&delta[(long)d * 3 + a], f * pdiff[e * 3 + a] * inv);
}

// per-batch mean-centering of delta, then fv_pos += delta - mean  (N==256)
__global__ void __launch_bounds__(256) k_pos_final(const float* __restrict__ pos,
                                                   const float* __restrict__ delta,
                                                   float* __restrict__ out) {
    __shared__ float s[3 * 256];
    int b = blockIdx.x, t = threadIdx.x;
    long base = ((long)b * NN + t) * 3;
    float d0 = delta[base + 0], d1 = delta[base + 1], d2 = delta[base + 2];
    s[0 * 256 + t] = d0; s[1 * 256 + t] = d1; s[2 * 256 + t] = d2;
    __syncthreads();
    for (int st = 128; st > 0; st >>= 1) {
        if (t < st)
            for (int a = 0; a < 3; ++a) s[a * 256 + t] += s[a * 256 + t + st];
        __syncthreads();
    }
    float m0 = s[0] * (1.0f / NN), m1 = s[256] * (1.0f / NN), m2 = s[512] * (1.0f / NN);
    out[base + 0] = pos[base + 0] + d0 - m0;
    out[base + 1] = pos[base + 1] + d1 - m1;
    out[base + 2] = pos[base + 2] + d2 - m2;
}

// ---------------------------------------------------------------------------
// Host side
// ---------------------------------------------------------------------------
static void gemm(hipStream_t st, int M, int K, int N, const float* A,
                 const float* W, const float* bias, float* out, int act) {
    GemmArgs g{A, W, bias, out, M, K, N};
    dim3 grid((M + 127) / 128, (N + 15) / 16);
    if (act) k_gemm<1><<<grid, 256, 0, st>>>(g);
    else     k_gemm<0><<<grid, 256, 0, st>>>(g);
}

// JAX pytree leaf order (dict keys sorted, None dropped): 67 leaves
static const int leaf_sizes[67] = {
    1024, 32, 128, 4,                               // e_geo_a_mlp
    2048, 64, 2048, 64, 2048, 64,                   // e_geo_readout
    96, 32, 1024, 32, 1024,                         // e_mlp_geo
    4096, 64, 4096, 64, 4096, 64,                   // e_readout
    14336, 64, 4096, 64, 8192, 128,                 // e_update_mlp_1
    4096, 64, 4096, 64,                             // e_update_mlp_2
    16384, 128, 512,                                // g_a_mlp
    32896, 128, 16384, 128,                         // g_update_mlp
    49152, 49152, 384, 384,                         // gru: W_hh, W_ih, b_hh, b_ih
    32768, 256,                                     // lin_g2v_update
    32768, 256,                                     // lin_v_update
    4096, 64, 4096, 64, 64,                         // mlp_pos_msg
    8192, 64,                                       // v2e_mul1
    8192, 64,                                       // v2e_mul2
    24576, 128, 16384, 128,                         // v_msg_mlp
    16384, 128, 16384, 128, 16384, 128,             // v_readout
    16384, 128                                      // v_update_proj
};
enum {
    L_EGA0W=0, L_EGA0B, L_EGA1W, L_EGA1B,
    L_EGR0W, L_EGR0B, L_EGR1W, L_EGR1B, L_EGR2W, L_EGR2B,
    L_EMG0W, L_EMG0B, L_EMG1W, L_EMG1B, L_EMG2W,
    L_ER0W, L_ER0B, L_ER1W, L_ER1B, L_ER2W, L_ER2B,
    L_EU10W, L_EU10B, L_EU11W, L_EU11B, L_EU12W, L_EU12B,
    L_EU20W, L_EU20B, L_EU21W, L_EU21B,
    L_GA0W, L_GA0B, L_GA1W,
    L_GU0W, L_GU0B, L_GU1W, L_GU1B,
    L_WHH, L_WIH, L_BHH, L_BIH,
    L_G2VW, L_G2VB,
    L_LVW, L_LVB,
    L_PM0W, L_PM0B, L_PM1W, L_PM1B, L_PM2W,
    L_M1W, L_M1B, L_M2W, L_M2B,                     // v2e_mul1, v2e_mul2
    L_VM0W, L_VM0B, L_VM1W, L_VM1B,
    L_VR0W, L_VR0B, L_VR1W, L_VR1B, L_VR2W, L_VR2B,
    L_VUPW, L_VUPB
};

extern "C" void kernel_launch(void* const* d_in, const int* in_sizes, int n_in,
                              void* d_out, int out_size, void* d_ws, size_t ws_size,
                              hipStream_t stream) {
    const float* fv     = (const float*)d_in[0];
    const float* fe     = (const float*)d_in[1];
    const float* fg     = (const float*)d_in[2];
    const float* fv_pos = (const float*)d_in[3];
    const int*   esrc   = (const int*)d_in[4];
    const int*   edst   = (const int*)d_in[5];

    const float* P[67];
    if (n_in >= 6 + 67) {
        for (int i = 0; i < 67; ++i) P[i] = (const float*)d_in[6 + i];
    } else {
        const float* base = (const float*)d_in[6];
        size_t o = 0;
        for (int i = 0; i < 67; ++i) { P[i] = base + o; o += (size_t)leaf_sizes[i]; }
    }

    // output regions
    float* o_fv  = (float*)d_out;
    float* o_fe  = o_fv + (size_t)NV * DV;
    float* o_fg  = o_fe + (size_t)NE * DE;
    float* o_pos = o_fg + (size_t)BB * DG;

    // workspace bump allocator
    size_t woff = 0;
    auto alloc = [&](size_t n) { float* p = (float*)d_ws + woff; woff += n; return p; };

    float* fv_dist = alloc(NV);
    float* fe_dist = alloc(NE);
    float* pdiff   = alloc((size_t)NE * 3);
    float* geo_in  = alloc((size_t)NE * 3);
    float* geoA    = alloc((size_t)NE * DEG);       // scratch 32
    float* geoB    = alloc((size_t)NE * DEG);       // final fe_geo
    float* g2v     = alloc((size_t)BB * 256);
    float* fv_mod  = alloc((size_t)NV * DV);
    float* vu      = alloc((size_t)NV * 256);
    float* CAT     = alloc((size_t)NE * 224);       // msg_in(192) -> u1_in(224)
    float* R1      = alloc((size_t)NE * 128);       // m1 -> u1 -> h
    float* R2      = alloc((size_t)NE * 128);       // msg -> u2|fe_t -> fe_mid -> c1|c2
    float* ggeo    = alloc((size_t)NE * NH);
    float* gg_h    = alloc((size_t)BB * DG);
    float* gg      = alloc((size_t)BB * NH);
    float* fv_agg  = alloc((size_t)NV * DV);
    float* a1      = alloc((size_t)NV * DE);
    float* a2      = alloc((size_t)NV * DE);
    float* red_fe  = alloc((size_t)BB * 3 * DE);
    float* red_fv  = alloc((size_t)BB * 3 * DV);
    float* red_geo = alloc((size_t)BB * 3 * DEG);
    float* Wc_e    = alloc((size_t)3 * DE * DE);   float* bc_e = alloc(DE);
    float* Wc_v    = alloc((size_t)3 * DV * DG);   float* bc_v = alloc(DG);
    float* Wc_g    = alloc((size_t)3 * DEG * (DG/2)); float* bc_g = alloc(DG/2);
    float* fe2g    = alloc((size_t)BB * (DG/2));
    float* fv2g    = alloc((size_t)BB * DG);
    float* geo2g   = alloc((size_t)BB * (DG/2));
    float* xh_in   = alloc((size_t)BB * 257);
    float* xh      = alloc((size_t)BB * DG);
    float* xx      = alloc((size_t)BB * DG);
    float* gi      = alloc((size_t)BB * 3 * DG);
    float* gh      = alloc((size_t)BB * 3 * DG);
    float* coef    = alloc(NE);
    float* delta   = alloc((size_t)NV * 3);
    (void)ws_size; (void)in_sizes; (void)out_size;

    const int T = 256;
    auto blk = [](long n) { return (unsigned)((n + 255) / 256); };

    // ---- geometry prep --------------------------------------------------
    k_dist<<<blk(NV), T, 0, stream>>>(fv_pos, fv_dist, NV);
    k_edge_pre<<<blk(NE), T, 0, stream>>>(fv_pos, fv_dist, esrc, edst,
                                          pdiff, fe_dist, geo_in, NE);

    // ---- FiLM from graph state -----------------------------------------
    gemm(stream, BB, DG, 256, fg, P[L_G2VW], P[L_G2VB], g2v, 0);
    k_film<<<blk((long)NV * DV), T, 0, stream>>>(fv, g2v, fv_mod, NV * DV);

    // ---- fe_geo = mlp3(e_mlp_geo, [fe_dist, fv_dist[src], fv_dist[dst]])
    gemm(stream, NE, 3,   DEG, geo_in, P[L_EMG0W], P[L_EMG0B], geoA, 1);
    gemm(stream, NE, DEG, DEG, geoA,   P[L_EMG1W], P[L_EMG1B], R1,   1);
    gemm(stream, NE, DEG, DEG, R1,     P[L_EMG2W], nullptr,    geoB, 0);  // fe_geo

    // ---- vu = lin_v_update(fv_mod) -------------------------------------
    gemm(stream, NV, DV, 256, fv_mod, P[L_LVW], P[L_LVB], vu, 0);

    // ---- msg = gelu(mlp2(v_msg_mlp, [fv_mod[src], fe])) ----------------
    k_cat_msg<<<blk((long)NE * 48), T, 0, stream>>>(fv_mod, fe, esrc, CAT, NE * 48);
    gemm(stream, NE, 192, 128, CAT, P[L_VM0W], P[L_VM0B], R1, 1);
    gemm(stream, NE, 128, 128, R1,  P[L_VM1W], P[L_VM1B], R2, 1);         // msg in R2

    // ---- gates ----------------------------------------------------------
    gemm(stream, NE, DEG, DEG, geoB, P[L_EGA0W], P[L_EGA0B], geoA, 1);
    gemm(stream, NE, DEG, NH,  geoA, P[L_EGA1W], P[L_EGA1B], ggeo, 0);
    gemm(stream, BB, DG, DG, fg,   P[L_GA0W], P[L_GA0B], gg_h, 1);
    gemm(stream, BB, DG, NH, gg_h, P[L_GA1W], nullptr,   gg,   0);

    // ---- gated aggregation ---------------------------------------------
    hipMemsetAsync(fv_agg, 0, (size_t)NV * DV * sizeof(float), stream);
    k_msg_agg<<<blk((long)NE * DV), T, 0, stream>>>(R2, vu, ggeo, gg, edst, fv_agg,
                                                    NE * DV);

    // ---- fv = v_update_proj(agg)  -> output fv -------------------------
    gemm(stream, NV, DV, DV, fv_agg, P[L_VUPW], P[L_VUPB], o_fv, 0);

    // ---- ab = v2e_mul1(fv)[src] * v2e_mul2(fv)[dst] (folded into cat) --
    gemm(stream, NV, DV, DE, o_fv, P[L_M1W], P[L_M1B], a1, 0);
    gemm(stream, NV, DV, DE, o_fv, P[L_M2W], P[L_M2B], a2, 0);

    // ---- h = mlp3(e_update_mlp_1, [ab, fe_geo, fg_e]) ------------------
    k_cat_u1<<<blk((long)NE * 56), T, 0, stream>>>(a1, a2, geoB, fg, esrc, edst,
                                                   CAT, NE * 56);
    float* u1 = R1;                 // NE x 64 (m1 dead)
    float* u2 = R2;                 // NE x 64 (msg dead)
    gemm(stream, NE, 224, DE, CAT, P[L_EU10W], P[L_EU10B], u1, 1);
    gemm(stream, NE, DE,  DE, u1,  P[L_EU11W], P[L_EU11B], u2, 1);
    float* h = R1;                  // full NE x 128 (u1 dead)
    gemm(stream, NE, DE, 128, u2, P[L_EU12W], P[L_EU12B], h, 0);

    // ---- fe update ------------------------------------------------------
    float* fe_t   = R2 + (size_t)NE * DE;
    float* fe_mid = R2;
    gemm(stream, NE, DE, DE, fe,   P[L_EU20W], P[L_EU20B], fe_t,   1);
    gemm(stream, NE, DE, DE, fe_t, P[L_EU21W], P[L_EU21B], fe_mid, 0);
    k_fe_out<<<blk((long)NE * DE), T, 0, stream>>>(fe_mid, h, o_fe, NE * DE);

    // ---- readouts (mean|min|max concat + stacked-weight GEMM) ----------
    k_colstats<<<blk(BB * DE), T, 0, stream>>>(o_fe, EE, DE, red_fe, BB * DE);
    k_colstats<<<blk(BB * DV), T, 0, stream>>>(o_fv, NN, DV, red_fv, BB * DV);
    k_colstats<<<blk(BB * DEG), T, 0, stream>>>(geoB, EE, DEG, red_geo, BB * DEG);
    k_cat3<<<blk(3 * DE * DE), T, 0, stream>>>(P[L_ER0W], P[L_ER1W], P[L_ER2W],
                                               P[L_ER0B], P[L_ER1B], P[L_ER2B],
                                               DE, DE, Wc_e, bc_e);
    k_cat3<<<blk(3 * DV * DG), T, 0, stream>>>(P[L_VR0W], P[L_VR1W], P[L_VR2W],
                                               P[L_VR0B], P[L_VR1B], P[L_VR2B],
                                               DV, DG, Wc_v, bc_v);
    k_cat3<<<blk(3 * DEG * (DG/2)), T, 0, stream>>>(P[L_EGR0W], P[L_EGR1W], P[L_EGR2W],
                                                    P[L_EGR0B], P[L_EGR1B], P[L_EGR2B],
                                                    DEG, DG/2, Wc_g, bc_g);
    gemm(stream, BB, 3 * DE,  DE,   red_fe,  Wc_e, bc_e, fe2g,  0);
    gemm(stream, BB, 3 * DV,  DG,   red_fv,  Wc_v, bc_v, fv2g,  0);
    gemm(stream, BB, 3 * DEG, DG/2, red_geo, Wc_g, bc_g, geo2g, 0);

    // ---- graph update MLP + GRU ----------------------------------------
    k_cat_xh<<<blk(BB * 257), T, 0, stream>>>(fe2g, fv2g, geo2g, xh_in, BB * 257);
    gemm(stream, BB, 257, DG, xh_in, P[L_GU0W], P[L_GU0B], xh, 1);
    gemm(stream, BB, DG, DG, xh, P[L_GU1W], P[L_GU1B], xx, 0);
    gemm(stream, BB, DG, 3 * DG, xx, P[L_WIH], P[L_BIH], gi, 0);
    gemm(stream, BB, DG, 3 * DG, fg, P[L_WHH], P[L_BHH], gh, 0);
    k_gru<<<blk(BB * DG), T, 0, stream>>>(gi, gh, fg, o_fg, BB * DG);

    // ---- position message -----------------------------------------------
    float* c1 = R2;
    float* c2 = R2 + (size_t)NE * DE;
    gemm(stream, NE, DE, DE, o_fe, P[L_PM0W], P[L_PM0B], c1, 1);
    gemm(stream, NE, DE, DE, c1,   P[L_PM1W], P[L_PM1B], c2, 1);
    gemm(stream, NE, DE, 1,  c2,   P[L_PM2W], nullptr,   coef, 0);
    hipMemsetAsync(delta, 0, (size_t)NV * 3 * sizeof(float), stream);
    k_posmsg<<<blk(NE), T, 0, stream>>>(coef, pdiff, fe_dist, edst, delta, NE);
    k_pos_final<<<BB, NN, 0, stream>>>(fv_pos, delta, o_pos);
}
